// TRT_26242250179226
// MI455X (gfx1250) — matrix operand order repaired and checked by reading, not compiled, not run on hardware
//
#include <hip/hip_runtime.h>
#include <hip/hip_bf16.h>
#include <stdint.h>

typedef __bf16 bf16_t;
typedef __attribute__((ext_vector_type(16))) __bf16 v16bf;
typedef __attribute__((ext_vector_type(8)))  __bf16 v8bf;
typedef __attribute__((ext_vector_type(8)))  float  v8f;

#define NTOK 16384
#define DCH  768
#define FDIM 2304
#define H1   2048
#define H2   1024

// ---------------------------------------------------------------------------
// Phase 1: per-chunk channel sums (64 chunks x 768 channels)
// ---------------------------------------------------------------------------
__global__ __launch_bounds__(256)
void partial_sums(const float* __restrict__ h, float* __restrict__ bs)
{
    int c     = blockIdx.x * 256 + threadIdx.x;   // 0..767
    int chunk = blockIdx.y;                       // 0..63
    const float* base = h + (size_t)chunk * 256 * DCH + c;
    float s = 0.f;
    for (int r = 0; r < 256; ++r) s += base[(size_t)r * DCH];
    bs[chunk * DCH + c] = s;
}

// ---------------------------------------------------------------------------
// Phase 2: exclusive scan of chunk sums per channel + totals. 1 block, 768 thr
// ---------------------------------------------------------------------------
__global__ void scan_chunks(float* __restrict__ bs, float* __restrict__ total)
{
    int c = threadIdx.x;                          // 0..767
    float run = 0.f;
    for (int k = 0; k < 64; ++k) {
        float v = bs[k * DCH + c];
        bs[k * DCH + c] = run;                    // exclusive offset
        run += v;
    }
    total[c] = run;
}

// ---------------------------------------------------------------------------
// Phase 3: fused feature build -> bf16 feat (16384 x 2304)
// ---------------------------------------------------------------------------
__global__ __launch_bounds__(256)
void build_features(const float* __restrict__ h, const float* __restrict__ bs,
                    const float* __restrict__ total, bf16_t* __restrict__ feat)
{
    int c     = blockIdx.x * 256 + threadIdx.x;
    int chunk = blockIdx.y;
    float run = bs[chunk * DCH + c];
    float tot = total[c];
    for (int r = 0; r < 256; ++r) {
        int row = chunk * 256 + r;
        float v = h[(size_t)row * DCH + c];
        float excl = run;
        run += v;
        float pre = (row > 0) ? excl / (float)row : 0.f;
        int cnt = (NTOK - 1) - row;
        float suf = (cnt > 0) ? (tot - run) / (float)cnt : 0.f;
        size_t o = (size_t)row * FDIM;
        feat[o + c]            = (bf16_t)pre;
        feat[o + DCH + c]      = (bf16_t)v;
        feat[o + 2 * DCH + c]  = (bf16_t)suf;
    }
}

// ---------------------------------------------------------------------------
// Weight convert + transpose: (K x N) f32 -> (N x K) bf16. block (32,8)
// ---------------------------------------------------------------------------
__global__ void transpose_convert(const float* __restrict__ in, bf16_t* __restrict__ outT,
                                  int K, int N)
{
    __shared__ float tile[32][33];
    int x0 = blockIdx.x * 32;  // n
    int y0 = blockIdx.y * 32;  // k
    int tx = threadIdx.x, ty = threadIdx.y;
    for (int j = ty; j < 32; j += 8)
        tile[j][tx] = in[(size_t)(y0 + j) * N + (x0 + tx)];
    __syncthreads();
    for (int j = ty; j < 32; j += 8)
        outT[(size_t)(x0 + j) * K + (y0 + tx)] = (bf16_t)tile[tx][j];
}

// ---------------------------------------------------------------------------
// WMMA bf16 GEMM: C(bf16, MxN) = relu(A(MxK) * Bt(NxK)^T + bias)
// Block tile 256x128, BK=32, 8 waves (4x2), wave tile 64x64 (4x4 of 16x16).
// LDS double-buffered: one barrier per K-step.
// ---------------------------------------------------------------------------
#define BM 256
#define BN 128
#define BK 32
#define LDT 40   // padded LDS K-stride (elements); 80B rows, 16B aligned, no bank conflicts

__global__ __launch_bounds__(256)
void gemm_bf16_wmma(const bf16_t* __restrict__ A, const bf16_t* __restrict__ Bt,
                    const float* __restrict__ bias, bf16_t* __restrict__ C,
                    int K, int N, int do_relu)
{
    __shared__ __align__(16) bf16_t sA[2][BM * LDT];
    __shared__ __align__(16) bf16_t sB[2][BN * LDT];

    const int tid  = threadIdx.x;
    const int lane = tid & 31;
    const int wave = tid >> 5;       // 0..7
    const int wm   = wave >> 1;      // 0..3 : 64-row slab
    const int wn   = wave & 1;       // 0..1 : 64-col slab
    const int m0   = blockIdx.y * BM;
    const int n0   = blockIdx.x * BN;

    const int lhr = lane & 15;          // row/col within 16x16 tile
    const int kbA = (lane >> 4) * 8;    // unused; A frag also reads at kbB (contiguous)
    const int kbB = (lane >> 4) * 16;   // B frag: contiguous 16-K run

    // one A row (32 bf16 = 64B) per thread; one B row for threads < 128
    const bf16_t* gA = A + (size_t)(m0 + tid) * K;
    const bf16_t* gB = Bt + (size_t)(n0 + (tid & (BN - 1))) * K;
    const bool hasB = (tid < BN);

    v8f  acc[4][4] = {};
    v8bf ra[4], rb[4];

    auto gload = [&](int k0) {
        const bf16_t* pa = gA + k0;
        ra[0] = *(const v8bf*)(pa);
        ra[1] = *(const v8bf*)(pa + 8);
        ra[2] = *(const v8bf*)(pa + 16);
        ra[3] = *(const v8bf*)(pa + 24);
        if (hasB) {
            const bf16_t* pb = gB + k0;
            rb[0] = *(const v8bf*)(pb);
            rb[1] = *(const v8bf*)(pb + 8);
            rb[2] = *(const v8bf*)(pb + 16);
            rb[3] = *(const v8bf*)(pb + 24);
        }
    };
    auto lstore = [&](int b) {
        bf16_t* da = &sA[b][tid * LDT];
        *(v8bf*)(da)      = ra[0];
        *(v8bf*)(da + 8)  = ra[1];
        *(v8bf*)(da + 16) = ra[2];
        *(v8bf*)(da + 24) = ra[3];
        if (hasB) {
            bf16_t* db = &sB[b][tid * LDT];
            *(v8bf*)(db)      = rb[0];
            *(v8bf*)(db + 8)  = rb[1];
            *(v8bf*)(db + 16) = rb[2];
            *(v8bf*)(db + 24) = rb[3];
        }
    };
    auto compute_tile = [&](int b) {
        union Frag { v16bf v; v8bf h[2]; };
        Frag af[4], bfr[4];
        const bf16_t* pA = &sA[b][0];
        const bf16_t* pB = &sB[b][0];
        #pragma unroll
        for (int mt = 0; mt < 4; ++mt) {
            int row = wm * 64 + mt * 16 + lhr;
            af[mt].h[0] = *(const v8bf*)&pA[row * LDT + kbB];        // K 0..7 / 16..23
            af[mt].h[1] = *(const v8bf*)&pA[row * LDT + kbB + 8];    // K 8..15 / 24..31
        }
        #pragma unroll
        for (int nt = 0; nt < 4; ++nt) {
            int col = wn * 64 + nt * 16 + lhr;
            bfr[nt].h[0] = *(const v8bf*)&pB[col * LDT + kbB];       // K 0..15 / 16..31
            bfr[nt].h[1] = *(const v8bf*)&pB[col * LDT + kbB + 8];
        }
        #pragma unroll
        for (int mt = 0; mt < 4; ++mt)
            #pragma unroll
            for (int nt = 0; nt < 4; ++nt)
                acc[mt][nt] = __builtin_amdgcn_wmma_f32_16x16x32_bf16(
                    false, af[mt].v, false, bfr[nt].v,
                    (short)0, acc[mt][nt], false, false);
    };

    // prologue: tile 0 -> buf 0
    gload(0);
    lstore(0);
    __syncthreads();

    int buf = 0;
    for (int k0 = BK; k0 < K; k0 += BK) {
        gload(k0);             // fetch tile k0 (overlaps with compute below)
        compute_tile(buf);     // consume previous tile
        lstore(buf ^ 1);       // fill other buffer
        __syncthreads();       // single barrier per K-step
        buf ^= 1;
    }
    compute_tile(buf);         // last tile

    // epilogue: bias + relu + bf16 store. C layout: lanes16-31 hold M=8..15
    const int rbase = (lane >> 4) * 8;
    #pragma unroll
    for (int mt = 0; mt < 4; ++mt) {
        #pragma unroll
        for (int nt = 0; nt < 4; ++nt) {
            int col = n0 + wn * 64 + nt * 16 + lhr;
            float bv = bias ? bias[col] : 0.f;
            #pragma unroll
            for (int r = 0; r < 8; ++r) {
                int row = m0 + wm * 64 + mt * 16 + rbase + r;
                float v = acc[mt][nt][r] + bv;
                if (do_relu) v = v > 0.f ? v : 0.f;
                C[(size_t)row * N + col] = (bf16_t)v;
            }
        }
    }
}

// ---------------------------------------------------------------------------
// Final GEMV: out[m] = x2[m,:] . W3 + b3.  Wave per row, f32 output.
// ---------------------------------------------------------------------------
__global__ __launch_bounds__(256)
void final_dot(const bf16_t* __restrict__ x2, const float* __restrict__ W3,
               const float* __restrict__ b3, float* __restrict__ out)
{
    int lane = threadIdx.x & 31;
    int row  = blockIdx.x * 8 + (threadIdx.x >> 5);
    const bf16_t* xr = x2 + (size_t)row * H2;
    float s = 0.f;
    for (int i = lane; i < H2; i += 32)
        s += (float)xr[i] * W3[i];
    #pragma unroll
    for (int off = 16; off > 0; off >>= 1)
        s += __shfl_xor(s, off, 32);
    if (lane == 0) out[row] = s + b3[0];
}

// ---------------------------------------------------------------------------
extern "C" void kernel_launch(void* const* d_in, const int* in_sizes, int n_in,
                              void* d_out, int out_size, void* d_ws, size_t ws_size,
                              hipStream_t stream)
{
    const float* h  = (const float*)d_in[0];
    const float* W1 = (const float*)d_in[1];
    const float* b1 = (const float*)d_in[2];
    const float* W2 = (const float*)d_in[3];
    const float* b2 = (const float*)d_in[4];
    const float* W3 = (const float*)d_in[5];
    const float* b3 = (const float*)d_in[6];
    float* out = (float*)d_out;

    char* p = (char*)d_ws;
    auto carve = [&](size_t bytes) -> void* {
        void* r = (void*)p;
        p += (bytes + 255) & ~(size_t)255;
        return r;
    };
    float*  bs    = (float*) carve((size_t)64 * DCH * 4);
    float*  total = (float*) carve((size_t)DCH * 4);
    bf16_t* feat  = (bf16_t*)carve((size_t)NTOK * FDIM * 2);
    bf16_t* w1t   = (bf16_t*)carve((size_t)FDIM * H1 * 2);
    bf16_t* w2t   = (bf16_t*)carve((size_t)H1 * H2 * 2);
    bf16_t* x1    = (bf16_t*)carve((size_t)NTOK * H1 * 2);
    bf16_t* x2    = (bf16_t*)carve((size_t)NTOK * H2 * 2);

    // features
    partial_sums  <<<dim3(3, 64), 256, 0, stream>>>(h, bs);
    scan_chunks   <<<1, DCH, 0, stream>>>(bs, total);
    build_features<<<dim3(3, 64), 256, 0, stream>>>(h, bs, total, feat);

    // weight convert + transpose to (N,K) bf16
    transpose_convert<<<dim3(H1 / 32, FDIM / 32), dim3(32, 8), 0, stream>>>(W1, w1t, FDIM, H1);
    transpose_convert<<<dim3(H2 / 32, H1 / 32),   dim3(32, 8), 0, stream>>>(W2, w2t, H1, H2);

    // GEMM1: x1 = relu(feat @ W1 + b1)   (16384x2304)*(2304x2048)
    gemm_bf16_wmma<<<dim3(H1 / BN, NTOK / BM), 256, 0, stream>>>(feat, w1t, b1, x1, FDIM, H1, 1);
    // GEMM2: x2 = relu(x1 @ W2 + b2)     (16384x2048)*(2048x1024)
    gemm_bf16_wmma<<<dim3(H2 / BN, NTOK / BM), 256, 0, stream>>>(x1, w2t, b2, x2, H1, H2, 1);
    // out = x2 @ W3 + b3
    final_dot<<<NTOK / 8, 256, 0, stream>>>(x2, W3, b3, out);
}
